// CapsuleNet_14053132993113
// MI455X (gfx1250) — compile-verified, matrix operand order checked by reading, not run on hardware
//
#include <hip/hip_runtime.h>
#include <hip/hip_bf16.h>
#include <stdint.h>

// ---------------------------------------------------------------------------
// CapsuleNet forward for MI455X (gfx1250, wave32, WMMA + TDM).
// pconv implicit GEMM: A (im2col) via vector loads, B (weights) via
// tensor_load_to_lds (TDM, TENSORcnt), compute on v_wmma_f32_16x16x32_bf16.
// ---------------------------------------------------------------------------

typedef __attribute__((ext_vector_type(16))) __bf16 v16bf;
typedef __attribute__((ext_vector_type(8)))  float  v8f;
typedef __attribute__((ext_vector_type(4)))  unsigned int v4u;
typedef __attribute__((ext_vector_type(8)))  int v8i;
typedef __attribute__((ext_vector_type(4)))  int v4i;

#define BATCH 16
#define F1C   256
#define F2C   256
#define IMGH  60
#define C1H   52      // conv1 output H/W
#define MAPD  22      // pconv output H/W
#define CAPD  8
#define NCMAP 32
#define PCAPS (NCMAP*MAPD*MAPD)   // 15488
#define NCLS  10
#define DDIM  16
#define H1D   512
#define H2D   1024
#define KPC   (81*256)            // 20736 (im2col K of pconv)

static __device__ __forceinline__ unsigned short f2bf(float f) {
  unsigned int u = __float_as_uint(f);
  u += 0x7FFFu + ((u >> 16) & 1u);          // round-to-nearest-even
  return (unsigned short)(u >> 16);
}

union Frag { uint4 u4[2]; v16bf v; };

// ------------------------------ conv1 (VALU) -------------------------------
__global__ void conv1_relu_bf16(const float* __restrict__ img,
                                const float* __restrict__ w,
                                const float* __restrict__ bias,
                                unsigned short* __restrict__ x1) {
  int idx = blockIdx.x * 256 + threadIdx.x;         // exact grid
  int c = idx & 255;
  int t = idx >> 8;
  int x = t % C1H; t /= C1H;
  int y = t % C1H; int b = t / C1H;
  const float* wp = w + c * 81;
  const float* ip = img + (b * IMGH + y) * IMGH + x;
  float acc = bias[c];
#pragma unroll
  for (int ky = 0; ky < 9; ++ky)
#pragma unroll
    for (int kx = 0; kx < 9; ++kx)
      acc = fmaf(ip[ky * IMGH + kx], wp[ky * 9 + kx], acc);
  x1[idx] = f2bf(fmaxf(acc, 0.f));
}

// ------------------------- weight repack kernels ---------------------------
__global__ void repack_pconv_w(const float* __restrict__ w,
                               unsigned short* __restrict__ wt) {
  int idx = blockIdx.x * 256 + threadIdx.x;         // exact grid
  int f2 = idx / KPC;
  int r  = idx - f2 * KPC;
  int pos = r >> 8;
  int c   = r & 255;
  wt[idx] = f2bf(w[(f2 * 256 + c) * 81 + pos]);
}

__global__ void transpose_to_bf16(const float* __restrict__ in,
                                  unsigned short* __restrict__ out,
                                  int K, int N) {
  int idx = blockIdx.x * 256 + threadIdx.x;
  if (idx >= K * N) return;
  int n = idx / K, k = idx - n * K;
  out[idx] = f2bf(in[k * N + n]);
}

// ------------------- pconv implicit GEMM (WMMA + TDM) ----------------------
__global__ void __launch_bounds__(256)
pconv_gemm_wmma(const unsigned short* __restrict__ x1,
                const unsigned short* __restrict__ wt,
                const float* __restrict__ bias,
                float* __restrict__ u_raw) {
  __shared__ unsigned short As[64][72];   // row stride 144B (16B aligned)
  __shared__ unsigned short Bs[64][72];   // [n][k], rows padded 128B + 16B

  const int tid  = threadIdx.x;
  const int lane = tid & 31;
  const int wid  = tid >> 5;
  const int wm   = wid & 3;               // 4 M sub-tiles of 16
  const int wn   = wid >> 2;              // 2 N sub-tiles of 32
  const int bm   = blockIdx.y * 64;       // 121 tiles (exact)
  const int bn   = blockIdx.x * 64;       // 4 tiles   (exact)

  // cooperative A-tile load coords: 64 rows x 4 segments of 16 bf16
  const int lr   = tid >> 2;
  const int lseg = tid & 3;
  const int gm   = bm + lr;
  const int bb   = gm / 484;
  const int rem  = gm - bb * 484;
  const int oy   = rem / MAPD;
  const int ox   = rem - oy * MAPD;

  // TDM destination: LDS byte offset of Bs (flat LDS addr low 32 bits)
  const unsigned int ldsB = (unsigned int)(size_t)(&Bs[0][0]);

  v8f c0 = {0,0,0,0,0,0,0,0};
  v8f c1 = {0,0,0,0,0,0,0,0};

  const int mrow = wm * 16 + (lane & 15);
  const int half = lane >> 4;
  const int n0   = wn * 32 + (lane & 15);

  for (int k0 = 0; k0 < KPC; k0 += 64) {
    // -------- B tile: TDM async tensor load (wave 0 issues + waits) --------
    if (wid == 0) {
      unsigned long long ga =
          (unsigned long long)(const void*)(wt + (size_t)bn * KPC + k0);
      v4u g0;
      g0.x = 1u;                                  // count=1, user descriptor
      g0.y = ldsB;                                // lds_addr
      g0.z = (unsigned int)ga;                    // global_addr[31:0]
      g0.w = ((unsigned int)(ga >> 32) & 0x01FFFFFFu) | (2u << 30); // type=2
      v8i g1;
      g1[0] = (int)((1u << 16)                    // data_size = 2 bytes
                  | (1u << 20)                    // pad_enable
                  | (4u << 22)                    // pad_interval = 32 DWORDs
                  | (3u << 25));                  // pad_amount   = 4 DWORDs
      g1[1] = (int)((unsigned int)(KPC & 0xFFFF) << 16);  // tensor_dim0 lo16
      g1[2] = (int)((KPC >> 16) | (64u << 16));   // dim0 hi16 | tensor_dim1 lo16
      g1[3] = (int)(64u << 16);                   // dim1 hi16(0) | tile_dim0=64
      g1[4] = 64;                                 // tile_dim1=64, tile_dim2=0
      g1[5] = KPC;                                // tensor_dim0_stride lo32
      g1[6] = 0;                                  // stride hi16 | dim1_stride lo
      g1[7] = 0;
      v4i gz4 = {0,0,0,0};
      v8i gz8 = {0,0,0,0,0,0,0,0};
      __builtin_amdgcn_tensor_load_to_lds(g0, g1, gz4, gz4, gz8, 0);
    }

    // -------- A tile: im2col gather (irregular rows -> vector path) --------
    const int pos   = k0 >> 8;                    // (ky,kx) const per chunk
    const int cbase = k0 & 255;
    const int ky = pos / 9, kx = pos - ky * 9;
    const int iy = 2 * oy + ky, ix = 2 * ox + kx;
    const unsigned short* ap =
        x1 + ((size_t)((bb * C1H + iy) * C1H + ix) << 8) + cbase + lseg * 16;
    uint4 av0 = *(const uint4*)(ap);
    uint4 av1 = *(const uint4*)(ap + 8);
    __builtin_prefetch(ap + 64, 0, 1);            // global_prefetch_b8
    *(uint4*)&As[lr][lseg * 16]     = av0;
    *(uint4*)&As[lr][lseg * 16 + 8] = av1;

    if (wid == 0) __builtin_amdgcn_s_wait_tensorcnt(0);
    __syncthreads();                              // publishes As + TDM Bs

#pragma unroll
    for (int ks = 0; ks < 64; ks += 32) {
      Frag a, b0f, b1f;
      // ISA 16-bit A layout: lanes 0-15 K{0-7,16-23}, lanes 16-31 K{8-15,24-31}
      a.u4[0]   = *(const uint4*)&As[mrow][ks + half * 8];
      a.u4[1]   = *(const uint4*)&As[mrow][ks + 16 + half * 8];
      b0f.u4[0] = *(const uint4*)&Bs[n0][ks + half * 8];
      b0f.u4[1] = *(const uint4*)&Bs[n0][ks + 16 + half * 8];
      b1f.u4[0] = *(const uint4*)&Bs[n0 + 16][ks + half * 8];
      b1f.u4[1] = *(const uint4*)&Bs[n0 + 16][ks + 16 + half * 8];
      c0 = __builtin_amdgcn_wmma_f32_16x16x32_bf16(false, a.v, false, b0f.v,
                                                   (short)0, c0, false, false);
      c1 = __builtin_amdgcn_wmma_f32_16x16x32_bf16(false, a.v, false, b1f.v,
                                                   (short)0, c1, false, false);
    }
    __syncthreads();
  }

  // epilogue: C/D layout lanes 0-15 N=lane M=r ; lanes 16-31 N=lane-16 M=r+8
  const int nl = lane & 15;
#pragma unroll
  for (int t2 = 0; t2 < 2; ++t2) {
    v8f cc = t2 ? c1 : c0;
    int f2 = bn + wn * 32 + t2 * 16 + nl;
    float bv = bias[f2];
    int e = f2 >> 5, mc = f2 & 31;
#pragma unroll
    for (int r = 0; r < 8; ++r) {
      int m  = bm + wm * 16 + r + half * 8;
      int b2 = m / 484;
      int rm = m - b2 * 484;
      int oy2 = rm / MAPD, ox2 = rm - oy2 * MAPD;
      int p = mc * 484 + oy2 * MAPD + ox2;
      u_raw[((size_t)(b2 * PCAPS + p) << 3) + e] = cc[r] + bv;
    }
  }
}

// ------------------------------ squash u -----------------------------------
__global__ void squash8(float* __restrict__ u) {
  int idx = blockIdx.x * 256 + threadIdx.x;
  if (idx >= BATCH * PCAPS) return;
  float* up = u + ((size_t)idx << 3);
  float s[8], sq = 0.f;
#pragma unroll
  for (int e = 0; e < 8; ++e) { s[e] = up[e]; sq = fmaf(s[e], s[e], sq); }
  float scale = (sq / (1.f + sq)) * rsqrtf(sq + 1e-8f);
#pragma unroll
  for (int e = 0; e < 8; ++e) up[e] = s[e] * scale;
}

__global__ void fill_zero(float* p, int n) {
  int i = blockIdx.x * 256 + threadIdx.x;
  if (i < n) p[i] = 0.f;
}

// -------------------------- routing kernels --------------------------------
__global__ void softmax_p(const float* __restrict__ bij, float* __restrict__ cij) {
  int bc = blockIdx.x;
  int b = bc / NCLS, c = bc - b * NCLS;
  const float* src = bij + (size_t)b * PCAPS * NCLS + c;
  float* dst       = cij + (size_t)b * PCAPS * NCLS + c;
  __shared__ float red[256];
  int tid = threadIdx.x;
  float mx = -1e30f;
  for (int p = tid; p < PCAPS; p += 256) mx = fmaxf(mx, src[(size_t)p * NCLS]);
  red[tid] = mx; __syncthreads();
  for (int s = 128; s > 0; s >>= 1) {
    if (tid < s) red[tid] = fmaxf(red[tid], red[tid + s]);
    __syncthreads();
  }
  mx = red[0]; __syncthreads();
  float sum = 0.f;
  for (int p = tid; p < PCAPS; p += 256) sum += __expf(src[(size_t)p * NCLS] - mx);
  red[tid] = sum; __syncthreads();
  for (int s = 128; s > 0; s >>= 1) {
    if (tid < s) red[tid] += red[tid + s];
    __syncthreads();
  }
  float inv = 1.f / red[0];
  for (int p = tid; p < PCAPS; p += 256)
    dst[(size_t)p * NCLS] = __expf(src[(size_t)p * NCLS] - mx) * inv;
}

__global__ void t_reduce(const float* __restrict__ cij, const float* __restrict__ u,
                         float* __restrict__ t) {
  int bc = blockIdx.x;
  int b = bc / NCLS, c = bc - b * NCLS;
  int tid = threadIdx.x;
  float acc[8] = {0,0,0,0,0,0,0,0};
  for (int p = tid; p < PCAPS; p += 256) {
    float cv = cij[((size_t)(b * PCAPS + p)) * NCLS + c];
    const float* up = u + ((size_t)(b * PCAPS + p) << 3);
#pragma unroll
    for (int e = 0; e < 8; ++e) acc[e] = fmaf(cv, up[e], acc[e]);
  }
  __shared__ float red[256 * 8];
#pragma unroll
  for (int e = 0; e < 8; ++e) red[tid * 8 + e] = acc[e];
  __syncthreads();
  for (int s = 128; s > 0; s >>= 1) {
    if (tid < s)
#pragma unroll
      for (int e = 0; e < 8; ++e) red[tid * 8 + e] += red[(tid + s) * 8 + e];
    __syncthreads();
  }
  if (tid < 8) t[bc * 8 + tid] = red[tid];
}

__global__ void route_small(const float* __restrict__ W, const float* __restrict__ t,
                            float* __restrict__ v, float* __restrict__ g) {
  int tid = threadIdx.x;
  if (tid >= BATCH * NCLS) return;
  int c = tid % NCLS;
  float te[8];
#pragma unroll
  for (int e = 0; e < 8; ++e) te[e] = t[tid * 8 + e];
  float sd[16], sq = 0.f;
#pragma unroll
  for (int d = 0; d < 16; ++d) {
    float s = 0.f;
#pragma unroll
    for (int e = 0; e < 8; ++e) s = fmaf(W[(c * 16 + d) * 8 + e], te[e], s);
    sd[d] = s; sq = fmaf(s, s, sq);
  }
  float scale = (sq / (1.f + sq)) * rsqrtf(sq + 1e-8f);
  float vd[16];
#pragma unroll
  for (int d = 0; d < 16; ++d) { vd[d] = sd[d] * scale; v[tid * 16 + d] = vd[d]; }
#pragma unroll
  for (int e = 0; e < 8; ++e) {
    float s = 0.f;
#pragma unroll
    for (int d = 0; d < 16; ++d) s = fmaf(W[(c * 16 + d) * 8 + e], vd[d], s);
    g[tid * 8 + e] = s;
  }
}

__global__ void b_update(const float* __restrict__ u, const float* __restrict__ g,
                         float* __restrict__ bij) {
  __shared__ float gs[BATCH * NCLS * 8];
  for (int i = threadIdx.x; i < BATCH * NCLS * 8; i += 256) gs[i] = g[i];
  __syncthreads();
  int idx = blockIdx.x * 256 + threadIdx.x;
  if (idx >= BATCH * PCAPS) return;
  int b = idx / PCAPS;
  const float* up = u + ((size_t)idx << 3);
  float ue[8];
#pragma unroll
  for (int e = 0; e < 8; ++e) ue[e] = up[e];
  float* bp = bij + (size_t)idx * NCLS;
#pragma unroll
  for (int c = 0; c < NCLS; ++c) {
    float s = 0.f;
#pragma unroll
    for (int e = 0; e < 8; ++e) s = fmaf(ue[e], gs[(b * NCLS + c) * 8 + e], s);
    bp[c] += s;
  }
}

// ---------------- v_length / argmax / one-hot / mask -----------------------
__global__ void mask_out(const float* __restrict__ v, float* __restrict__ out_vlen,
                         int* __restrict__ out_ohe, unsigned short* __restrict__ mbf) {
  __shared__ float vl[BATCH * NCLS];
  __shared__ int amax[BATCH];
  int tid = threadIdx.x;
  if (tid < BATCH * NCLS) {
    float s = 0.f;
#pragma unroll
    for (int d = 0; d < 16; ++d) s = fmaf(v[tid * 16 + d], v[tid * 16 + d], s);
    vl[tid] = sqrtf(s);
    out_vlen[tid] = vl[tid];
  }
  __syncthreads();
  if (tid < BATCH) {
    int best = 0; float bv = vl[tid * NCLS];
    for (int c = 1; c < NCLS; ++c)
      if (vl[tid * NCLS + c] > bv) { bv = vl[tid * NCLS + c]; best = c; }
    amax[tid] = best;
  }
  __syncthreads();
  if (tid < BATCH * NCLS) {
    int b = tid / NCLS, c = tid - b * NCLS;
    out_ohe[tid] = (c == amax[b]) ? 1 : 0;
  }
  for (int i = tid; i < BATCH * NCLS * DDIM; i += 256) {
    int b = i / (NCLS * DDIM);
    int j = i - b * (NCLS * DDIM);
    int c = j >> 4;
    mbf[i] = f2bf((c == amax[b]) ? v[i] : 0.f);
  }
}

// ----------------- decoder GEMM: M=16 WMMA, act epilogue -------------------
__global__ void __launch_bounds__(128)
gemm16_wmma(const unsigned short* __restrict__ A, const unsigned short* __restrict__ Bt,
            const float* __restrict__ bias, float* __restrict__ outF,
            unsigned short* __restrict__ outB, int K, int N, int act) {
  extern __shared__ unsigned short sA[];            // 16*K bf16
  int tid = threadIdx.x;
  int lane = tid & 31, wid = tid >> 5;
  int nuint = 8 * K;                                // 16*K/2
  for (int i = tid; i < nuint; i += 128)
    ((unsigned int*)sA)[i] = ((const unsigned int*)A)[i];
  __syncthreads();

  int nbase = blockIdx.x * 64 + wid * 16;
  int nrow  = nbase + (lane & 15);
  int ncl   = nrow < N ? nrow : (N - 1);
  const unsigned short* brow = Bt + (size_t)ncl * K;
  int half = lane >> 4;
  int mrow = lane & 15;
  v8f acc = {0,0,0,0,0,0,0,0};
  for (int k0 = 0; k0 < K; k0 += 32) {
    Frag a, b;
    a.u4[0] = *(const uint4*)&sA[mrow * K + k0 + half * 8];
    a.u4[1] = *(const uint4*)&sA[mrow * K + k0 + 16 + half * 8];
    b.u4[0] = *(const uint4*)(brow + k0 + half * 8);
    b.u4[1] = *(const uint4*)(brow + k0 + 16 + half * 8);
    acc = __builtin_amdgcn_wmma_f32_16x16x32_bf16(false, a.v, false, b.v,
                                                  (short)0, acc, false, false);
  }
  if (nrow < N) {
    float bv = bias[nrow];
#pragma unroll
    for (int r = 0; r < 8; ++r) {
      int m = r + half * 8;
      float val = acc[r] + bv;
      val = (act == 0) ? fmaxf(val, 0.f) : (1.f / (1.f + __expf(-val)));
      if (outF) outF[(size_t)m * N + nrow] = val;
      if (outB) outB[(size_t)m * N + nrow] = f2bf(val);
    }
  }
}

// ------------------------------- launcher ----------------------------------
extern "C" void kernel_launch(void* const* d_in, const int* in_sizes, int n_in,
                              void* d_out, int out_size, void* d_ws, size_t ws_size,
                              hipStream_t stream) {
  (void)in_sizes; (void)n_in; (void)out_size; (void)ws_size;
  const float* imgs = (const float*)d_in[0];
  const float* c1w  = (const float*)d_in[1];
  const float* c1b  = (const float*)d_in[2];
  const float* pw   = (const float*)d_in[3];
  const float* pb   = (const float*)d_in[4];
  const float* Wd   = (const float*)d_in[5];
  const float* w1   = (const float*)d_in[6];
  const float* b1   = (const float*)d_in[7];
  const float* w2   = (const float*)d_in[8];
  const float* b2   = (const float*)d_in[9];
  const float* w3   = (const float*)d_in[10];
  const float* b3   = (const float*)d_in[11];

  char* ws = (char*)d_ws;
  size_t off = 0;
  auto alloc = [&](size_t bytes) -> void* {
    void* p = (void*)(ws + off);
    off += (bytes + 255) & ~(size_t)255;
    return p;
  };

  unsigned short* x1  = (unsigned short*)alloc((size_t)BATCH*C1H*C1H*F1C * 2);
  unsigned short* wt  = (unsigned short*)alloc((size_t)KPC*F2C * 2);
  float* u   = (float*)alloc((size_t)BATCH*PCAPS*CAPD * 4);
  float* bij = (float*)alloc((size_t)BATCH*PCAPS*NCLS * 4);
  float* cij = (float*)alloc((size_t)BATCH*PCAPS*NCLS * 4);
  float* tb  = (float*)alloc((size_t)BATCH*NCLS*CAPD * 4);
  float* vb  = (float*)alloc((size_t)BATCH*NCLS*DDIM * 4);
  float* gb  = (float*)alloc((size_t)BATCH*NCLS*CAPD * 4);
  unsigned short* mbf = (unsigned short*)alloc((size_t)BATCH*NCLS*DDIM * 2);
  unsigned short* w1t = (unsigned short*)alloc((size_t)H1D*160 * 2);
  unsigned short* w2t = (unsigned short*)alloc((size_t)H2D*H1D * 2);
  unsigned short* w3t = (unsigned short*)alloc((size_t)3600*H2D * 2);
  unsigned short* h1  = (unsigned short*)alloc((size_t)BATCH*H1D * 2);
  unsigned short* h2  = (unsigned short*)alloc((size_t)BATCH*H2D * 2);

  // stage 1: conv1 + weight repacks
  conv1_relu_bf16<<<(BATCH*C1H*C1H*F1C)/256, 256, 0, stream>>>(imgs, c1w, c1b, x1);
  repack_pconv_w<<<(KPC*F2C)/256, 256, 0, stream>>>(pw, wt);
  transpose_to_bf16<<<(160*H1D + 255)/256, 256, 0, stream>>>(w1, w1t, 160, H1D);
  transpose_to_bf16<<<(H1D*H2D + 255)/256, 256, 0, stream>>>(w2, w2t, H1D, H2D);
  transpose_to_bf16<<<(H2D*3600 + 255)/256, 256, 0, stream>>>(w3, w3t, H2D, 3600);

  // stage 2: pconv implicit GEMM (dominant, WMMA + TDM)
  dim3 g2(4, 121);
  pconv_gemm_wmma<<<g2, 256, 0, stream>>>(x1, wt, pb, u);

  // stage 3: squash + routing (u_hat factored out, never materialized)
  squash8<<<(BATCH*PCAPS + 255)/256, 256, 0, stream>>>(u);
  fill_zero<<<(BATCH*PCAPS*NCLS + 255)/256, 256, 0, stream>>>(bij, BATCH*PCAPS*NCLS);
  for (int it = 0; it < 3; ++it) {
    softmax_p<<<BATCH*NCLS, 256, 0, stream>>>(bij, cij);
    t_reduce<<<BATCH*NCLS, 256, 0, stream>>>(cij, u, tb);
    route_small<<<1, 192, 0, stream>>>(Wd, tb, vb, gb);
    if (it < 2)
      b_update<<<(BATCH*PCAPS + 255)/256, 256, 0, stream>>>(u, gb, bij);
  }

  // stage 4: outputs + decoder (WMMA, M=16 exactly one tile)
  int*   ohe      = (int*)d_out;
  float* img_out  = (float*)d_out + BATCH*NCLS;
  float* vlen_out = (float*)d_out + BATCH*NCLS + BATCH*3600;
  mask_out<<<1, 256, 0, stream>>>(vb, vlen_out, ohe, mbf);
  gemm16_wmma<<<(H1D + 63)/64, 128, 16*160*2, stream>>>(mbf, w1t, b1, nullptr, h1, 160, H1D, 0);
  gemm16_wmma<<<(H2D + 63)/64, 128, 16*H1D*2, stream>>>(h1, w2t, b2, nullptr, h2, H1D, H2D, 0);
  gemm16_wmma<<<(3600 + 63)/64, 128, 16*H2D*2, stream>>>(h2, w3t, b3, img_out, nullptr, H2D, 3600, 1);
}
